// Classifier_1803886265511
// MI455X (gfx1250) — compile-verified, matrix operand order checked by reading, not run on hardware
//
#include <hip/hip_runtime.h>
#include <cstdint>
#include <cstddef>

// Problem constants (from reference): B=8192, H=768, L=2, k=3, G=5, NUM_COEF=8
#define B_ROWS 8192
#define IN_F   768
#define NCOEF  8
#define KDIM   (IN_F * 9)   // 6912 = augmented K (silu + 8 spline bases per input feature)
#define OUT_L  2

typedef __attribute__((ext_vector_type(16))) _Float16 v16h;
typedef __attribute__((ext_vector_type(8)))  _Float16 v8h;
typedef __attribute__((ext_vector_type(8)))  float    v8f;

__device__ __forceinline__ float silu_f(float x) {
  return x / (1.0f + __expf(-x));
}

// Order-3 B-spline bases on uniform extended knot grid t_j = -1 + (j-3)*0.4, j=0..11.
// Faithful Cox-de Boor recursion (denominators are r*h, never zero).
__device__ __forceinline__ void bspline8(float x, float* __restrict__ out) {
  const float h = 0.4f;
  float t[12];
#pragma unroll
  for (int j = 0; j < 12; ++j) t[j] = -1.0f + (float)(j - 3) * h;
  float b[11];
#pragma unroll
  for (int j = 0; j < 11; ++j) b[j] = (x >= t[j] && x < t[j + 1]) ? 1.0f : 0.0f;
#pragma unroll
  for (int r = 1; r <= 3; ++r) {
    const float inv = 1.0f / ((float)r * h);
#pragma unroll
    for (int j = 0; j < 11 - 3; ++j) { // only first 11-r needed; 8 suffices for final
      // general recursion valid for j < 11 - r; we keep full range via guard below
    }
#pragma unroll
    for (int j = 0; j < 10; ++j) {
      if (j < 11 - r)
        b[j] = (x - t[j]) * inv * b[j] + (t[j + r + 1] - x) * inv * b[j + 1];
    }
  }
#pragma unroll
  for (int g = 0; g < NCOEF; ++g) out[g] = b[g];
}

// ---------------- Prep kernels: build augmented f16 A and W^T ----------------

__global__ __launch_bounds__(256) void kan_prep_act(const float* __restrict__ x,
                                                    _Float16* __restrict__ A, int total) {
  int idx = blockIdx.x * blockDim.x + threadIdx.x;
  if (idx >= total) return;
  float v = x[idx];
  int i = idx % IN_F;
  int b = idx / IN_F;
  _Float16* p = A + (size_t)b * KDIM + (size_t)i * 9;
  p[0] = (_Float16)silu_f(v);
  float bs[NCOEF];
  bspline8(v, bs);
#pragma unroll
  for (int g = 0; g < NCOEF; ++g) p[1 + g] = (_Float16)bs[g];
}

__global__ __launch_bounds__(256) void kan_prep_w(const float* __restrict__ bw,
                                                  const float* __restrict__ sw,
                                                  _Float16* __restrict__ W, int total) {
  int idx = blockIdx.x * blockDim.x + threadIdx.x; // idx = o*IN_F + i
  if (idx >= total) return;
  int i = idx % IN_F;
  int o = idx / IN_F;
  _Float16* p = W + (size_t)o * KDIM + (size_t)i * 9;
  p[0] = (_Float16)bw[idx];
#pragma unroll
  for (int g = 0; g < NCOEF; ++g) p[1 + g] = (_Float16)sw[(size_t)idx * NCOEF + g];
}

// ---------------- WMMA GEMM: H = gelu(A[8192x6912] * W^T[6912x768]) ----------------

// 16-bit A-fragment layout (ISA 7.12.2): lane L<16 -> row M=L, K {0..7,16..23};
// lane L+16 -> row M=L, K {8..15,24..31}. Two contiguous 16B chunks per lane.
// Same pattern used for B from row-major [N][K] (transposed weight).
__device__ __forceinline__ v16h load_frag(const _Float16* __restrict__ base,
                                          int row0, int k0, int ldk, int lane) {
  const int row = row0 + (lane & 15);
  const int kk  = k0 + ((lane & 16) >> 1); // +8 halves for upper half-wave
  const _Float16* p = base + (size_t)row * ldk + kk;
  v8h lo = *(const v8h*)(p);
  v8h hi = *(const v8h*)(p + 16);
  v16h r;
#pragma unroll
  for (int t = 0; t < 8; ++t) { r[t] = lo[t]; r[t + 8] = hi[t]; }
  return r;
}

__global__ __launch_bounds__(128) void kan_gemm1_gelu(const _Float16* __restrict__ A,
                                                      const _Float16* __restrict__ W,
                                                      float* __restrict__ H) {
  const int lane = threadIdx.x & 31;
  const int wave = threadIdx.x >> 5;
  const int m0 = blockIdx.x * 64;                 // 4 M-fragments per wave
  const int n0 = blockIdx.y * 128 + wave * 32;    // 2 N-fragments per wave

  v8f acc[4][2] = {};

  for (int k0 = 0; k0 < KDIM; k0 += 32) {
    v16h af[4], bf[2];
#pragma unroll
    for (int m = 0; m < 4; ++m) af[m] = load_frag(A, m0 + 16 * m, k0, KDIM, lane);
#pragma unroll
    for (int n = 0; n < 2; ++n) bf[n] = load_frag(W, n0 + 16 * n, k0, KDIM, lane);
#pragma unroll
    for (int m = 0; m < 4; ++m)
#pragma unroll
      for (int n = 0; n < 2; ++n)
        acc[m][n] = __builtin_amdgcn_wmma_f32_16x16x32_f16(
            false, af[m], false, bf[n], (short)0, acc[m][n], false, false);
  }

  // D layout: lane L holds (m = v + 8*(L>=16), n = L&15) for v in 0..7
  const int rsub = (lane >> 4) << 3;
  const int csub = lane & 15;
#pragma unroll
  for (int m = 0; m < 4; ++m)
#pragma unroll
    for (int n = 0; n < 2; ++n)
#pragma unroll
      for (int v = 0; v < 8; ++v) {
        float x = acc[m][n][v];
        float g = 0.5f * x * (1.0f + erff(x * 0.70710678118654752f)); // exact-erf GELU
        int row = m0 + 16 * m + rsub + v;
        int col = n0 + 16 * n + csub;
        H[(size_t)row * IN_F + col] = g;
      }
}

// ---------------- Layer 2: out[b, 0..1], one wave per row, shuffle reduction ----------------

__global__ __launch_bounds__(256) void kan_layer2(const float* __restrict__ H,
                                                  const float* __restrict__ w2b,
                                                  const float* __restrict__ w2s,
                                                  float* __restrict__ out) {
  const int lane = threadIdx.x & 31;
  const int b = blockIdx.x * 8 + (threadIdx.x >> 5);
  float a0 = 0.0f, a1 = 0.0f;
  for (int i = lane; i < IN_F; i += 32) {
    float x = H[(size_t)b * IN_F + i];
    float s = silu_f(x);
    float bs[NCOEF];
    bspline8(x, bs);
    a0 += s * w2b[i];
    a1 += s * w2b[IN_F + i];
    const float* w0 = w2s + (size_t)i * NCOEF;
    const float* w1 = w2s + (size_t)(IN_F + i) * NCOEF;
#pragma unroll
    for (int g = 0; g < NCOEF; ++g) {
      a0 += bs[g] * w0[g];
      a1 += bs[g] * w1[g];
    }
  }
#pragma unroll
  for (int off = 16; off > 0; off >>= 1) {
    a0 += __shfl_down(a0, off, 32);
    a1 += __shfl_down(a1, off, 32);
  }
  if (lane == 0) {
    out[(size_t)b * OUT_L + 0] = a0;
    out[(size_t)b * OUT_L + 1] = a1;
  }
}

// ---------------- Launch ----------------

extern "C" void kernel_launch(void* const* d_in, const int* in_sizes, int n_in,
                              void* d_out, int out_size, void* d_ws, size_t ws_size,
                              hipStream_t stream) {
  (void)in_sizes; (void)n_in; (void)out_size; (void)ws_size;
  const float* hidden = (const float*)d_in[0];
  const float* kbw    = (const float*)d_in[1];  // [768,768]
  const float* ksw    = (const float*)d_in[2];  // [768,768,8]
  const float* obw    = (const float*)d_in[3];  // [2,768]
  const float* osw    = (const float*)d_in[4];  // [2,768,8]
  float* out = (float*)d_out;

  uint8_t* ws = (uint8_t*)d_ws;
  const size_t szA = (size_t)B_ROWS * KDIM * sizeof(_Float16); // 113.2 MB
  const size_t szW = (size_t)IN_F * KDIM * sizeof(_Float16);   // 10.6 MB
  _Float16* A = (_Float16*)(ws);
  _Float16* W = (_Float16*)(ws + szA);
  float*    H = (float*)(ws + szA + szW);                      // 25.2 MB

  const int totA = B_ROWS * IN_F;
  kan_prep_act<<<(totA + 255) / 256, 256, 0, stream>>>(hidden, A, totA);
  const int totW = IN_F * IN_F;
  kan_prep_w<<<(totW + 255) / 256, 256, 0, stream>>>(kbw, ksw, W, totW);

  dim3 g(B_ROWS / 64, IN_F / 128); // 128 x 6 blocks, 4 waves each
  kan_gemm1_gelu<<<g, 128, 0, stream>>>(A, W, H);

  kan_layer2<<<B_ROWS / 8, 256, 0, stream>>>(H, obw, osw, out);
}
